// TransformerBlock_37855841747489
// MI455X (gfx1250) — matrix operand order repaired and checked by reading, not compiled, not run on hardware
//
#include <hip/hip_runtime.h>
#include <hip/hip_bf16.h>
#include <cstdint>
#include <cstddef>

#define Bb   2
#define Dd   1024
#define Nn   2048
#define Hh   8
#define KVHh 2
#define DHh  64
#define FFf  2048
#define Mrows (Bb * Nn)   // 4096

typedef __attribute__((ext_vector_type(16))) __bf16 v16bf;
typedef __attribute__((ext_vector_type(8)))  float  v8f;

union FragBF { v16bf v; uint4 q[2]; };

__device__ __forceinline__ float bf2f(__bf16 b) {
  unsigned short h = __builtin_bit_cast(unsigned short, b);
  union { unsigned u; float f; } x; x.u = ((unsigned)h) << 16; return x.f;
}
__device__ __forceinline__ __bf16 f2bf(float f) {
  union { float f; unsigned u; } x; x.f = f;
  unsigned r = x.u + 0x7FFFu + ((x.u >> 16) & 1u);
  if ((x.u & 0x7F800000u) == 0x7F800000u) r = x.u;   // inf/nan passthrough
  return __builtin_bit_cast(__bf16, (unsigned short)(r >> 16));
}

// Async global -> LDS copy, 16 bytes per lane (GLOBAL_LOAD_ASYNC_TO_LDS_B128,
// tracked by ASYNCcnt; no destination VGPRs). LDS byte offset = low 32 bits of
// the generic shared-memory pointer (LDS aperture keeps offset in addr[31:0]).
__device__ __forceinline__ void g2lds_b128(const __bf16* g, __bf16* l) {
  asm volatile("global_load_async_to_lds_b128 %0, %1, off"
               :: "v"((unsigned)(uintptr_t)l), "v"(g)
               : "memory");
}

// ---------------------------------------------------------------------------
// Tiled transposes (layout prep)
// ---------------------------------------------------------------------------
__global__ void k_x_transpose(const float* __restrict__ x,
                              float* __restrict__ xtF, __bf16* __restrict__ xtB) {
  __shared__ float tile[32][33];
  const int b = blockIdx.z;
  const int nx = blockIdx.x * 32, dy = blockIdx.y * 32;
  const int tx = threadIdx.x, ty = threadIdx.y;
  const float* xb = x + (size_t)b * Dd * Nn;
#pragma unroll
  for (int j = 0; j < 32; j += 8)
    tile[ty + j][tx] = xb[(size_t)(dy + ty + j) * Nn + nx + tx];
  __syncthreads();
#pragma unroll
  for (int j = 0; j < 32; j += 8) {
    float v = tile[tx][ty + j];
    size_t o = (size_t)b * Nn * Dd + (size_t)(nx + ty + j) * Dd + dy + tx;
    xtF[o] = v;
    xtB[o] = f2bf(v);
  }
}

// W [K, Nc] f32  ->  Wt [Nc, K] bf16
__global__ void k_w_transpose(const float* __restrict__ W, __bf16* __restrict__ Wt,
                              int K, int Ncols) {
  __shared__ float tile[32][33];
  const int nx = blockIdx.x * 32, ky = blockIdx.y * 32;
  const int tx = threadIdx.x, ty = threadIdx.y;
#pragma unroll
  for (int j = 0; j < 32; j += 8)
    tile[ty + j][tx] = W[(size_t)(ky + ty + j) * Ncols + nx + tx];
  __syncthreads();
#pragma unroll
  for (int j = 0; j < 32; j += 8)
    Wt[(size_t)(nx + ty + j) * K + ky + tx] = f2bf(tile[tx][ty + j]);
}

// y [B,N,D] -> out [B,D,N]
__global__ void k_out_transpose(const float* __restrict__ y, float* __restrict__ o) {
  __shared__ float tile[32][33];
  const int b = blockIdx.z;
  const int dx = blockIdx.x * 32, ny = blockIdx.y * 32;
  const int tx = threadIdx.x, ty = threadIdx.y;
  const float* yb = y + (size_t)b * Nn * Dd;
#pragma unroll
  for (int j = 0; j < 32; j += 8)
    tile[ty + j][tx] = yb[(size_t)(ny + ty + j) * Dd + dx + tx];
  __syncthreads();
#pragma unroll
  for (int j = 0; j < 32; j += 8)
    o[(size_t)b * Dd * Nn + (size_t)(dx + ty + j) * Nn + ny + tx] = tile[tx][ty + j];
}

// ---------------------------------------------------------------------------
// Generic bf16 WMMA GEMM: C[M,Nc] = epilogue(A[M,K] @ Bt[Nc,K]^T)
// block tile 128x128, BK=32, 8 waves (2x4), each wave 64x32 (8 wmma accums).
// Tile staging uses async global->LDS (ASYNCcnt) with double buffering:
// prefetch tile kt+1 into buf[cur^1] while computing tile kt from buf[cur].
// s_wait_asynccnt 4 relies on in-order async-load completion (ISA 08 §4.1).
// Last iteration is peeled so the steady-state loop is branch-free.
// ---------------------------------------------------------------------------
__global__ __launch_bounds__(256)
void k_gemm_bf16(const __bf16* __restrict__ A, const __bf16* __restrict__ Bt,
                 int K, int Ncols,
                 const float* __restrict__ bias, int act,
                 const float* __restrict__ res,
                 float* __restrict__ outF, __bf16* __restrict__ outBF) {
  __shared__ __align__(16) __bf16 lds_a[2][128 * 40];
  __shared__ __align__(16) __bf16 lds_b[2][128 * 40];
  const int tid  = threadIdx.x;
  const int lane = tid & 31, w = tid >> 5;
  const int wm = w >> 2, wn = w & 3;        // 2 x 4 wave grid
  const int lm = lane & 15, lh = lane >> 4;
  const int bm = blockIdx.y * 128, bn = blockIdx.x * 128;
  const int ksteps = K >> 5;                // >= 16 for all calls here

  // this thread's staging slice: 2 x (row, 8 cols) per matrix per tile
  const int i0 = tid * 8,         r0 = i0 >> 5, c0 = i0 & 31;
  const int i1 = (tid + 256) * 8, r1 = i1 >> 5, c1 = i1 & 31;

  v8f zero;
#pragma unroll
  for (int e = 0; e < 8; ++e) zero[e] = 0.0f;
  v8f acc[8];
#pragma unroll
  for (int i = 0; i < 8; ++i) acc[i] = zero;

  auto stage_tile = [&](int k0, int buf) {
    g2lds_b128(A  + (size_t)(bm + r0) * K + k0 + c0, &lds_a[buf][r0 * 40 + c0]);
    g2lds_b128(A  + (size_t)(bm + r1) * K + k0 + c1, &lds_a[buf][r1 * 40 + c1]);
    g2lds_b128(Bt + (size_t)(bn + r0) * K + k0 + c0, &lds_b[buf][r0 * 40 + c0]);
    g2lds_b128(Bt + (size_t)(bn + r1) * K + k0 + c1, &lds_b[buf][r1 * 40 + c1]);
  };

  auto compute_tile = [&](int buf) {
    FragBF fa[4], fb[2];
#pragma unroll
    for (int m = 0; m < 4; ++m) {           // A frag: lane=row, K split by lane half
      int row = wm * 64 + m * 16 + lm;
      fa[m].q[0] = *(const uint4*)&lds_a[buf][row * 40 + lh * 8];
      fa[m].q[1] = *(const uint4*)&lds_a[buf][row * 40 + 16 + lh * 8];
    }
#pragma unroll
    for (int nt = 0; nt < 2; ++nt) {        // B frag: lane=col, K split by lane half
      int col = wn * 32 + nt * 16 + lm;
      fb[nt].q[0] = *(const uint4*)&lds_b[buf][col * 40 + lh * 8];
      fb[nt].q[1] = *(const uint4*)&lds_b[buf][col * 40 + 16 + lh * 8];
    }
#pragma unroll
    for (int m = 0; m < 4; ++m)
#pragma unroll
      for (int nt = 0; nt < 2; ++nt)
        acc[m * 2 + nt] = __builtin_amdgcn_wmma_f32_16x16x32_bf16(
            false, fa[m].v, false, fb[nt].v, (short)0, acc[m * 2 + nt], false, false);
  };

  // prologue: stage tile 0 into buffer 0 (4 async b128 ops per thread)
  stage_tile(0, 0);

  // steady state: branch-free (prefetch next, wait current, compute current)
  for (int kt = 0; kt < ksteps - 1; ++kt) {
    const int cur = kt & 1;
    __syncthreads();   // WAR: all waves done reading buf[cur^1] (tile kt-1)
    stage_tile((kt + 1) << 5, cur ^ 1);
    asm volatile("s_wait_asynccnt 0x4" ::: "memory");  // tile kt landed (in-order)
    __syncthreads();   // RAW: every wave's portion of tile kt is in LDS
    compute_tile(cur);
  }

  // peeled last tile
  __syncthreads();
  asm volatile("s_wait_asynccnt 0x0" ::: "memory");
  __syncthreads();
  compute_tile((ksteps - 1) & 1);

  // Epilogue: bias, activation, residual, f32 / bf16 stores
#pragma unroll
  for (int m = 0; m < 4; ++m) {
#pragma unroll
    for (int nt = 0; nt < 2; ++nt) {
      int colg = bn + wn * 32 + nt * 16 + lm;
      int rowb = bm + wm * 64 + m * 16 + lh * 8;
      float bi = bias ? bias[colg] : 0.0f;
#pragma unroll
      for (int r = 0; r < 8; ++r) {
        int rowg = rowb + r;
        float v = acc[m * 2 + nt][r] + bi;
        if (act == 1) v = v * (1.0f / (1.0f + __expf(-v)));   // SiLU
        size_t o = (size_t)rowg * Ncols + colg;
        if (res)   v += res[o];
        if (outF)  outF[o]  = v;
        if (outBF) outBF[o] = f2bf(v);
      }
    }
  }
}

// ---------------------------------------------------------------------------
// RMSNorm (per head) + xPos rotary; emits q,k bf16 and V^T bf16
// grid = B*N blocks of 256 (8 waves): wave w -> q head w; waves 0..1 -> k heads
// ---------------------------------------------------------------------------
__global__ __launch_bounds__(256)
void k_qkv_post(const float* __restrict__ qF, const float* __restrict__ kvF,
                const float* __restrict__ qg, const float* __restrict__ kg,
                __bf16* __restrict__ qh, __bf16* __restrict__ kh,
                __bf16* __restrict__ vT) {
  const int bn = blockIdx.x;
  const int b = bn / Nn, n = bn % Nn;
  const int tid = threadIdx.x;
  const int w = tid >> 5, lane = tid & 31;     // lane = rotary pair index (dh/2)

  const float i2 = 2.0f * lane;
  const float inv_freq = __expf(-(i2 / (float)DHh) * __logf(10000.0f));
  const float fr = (float)n * inv_freq;
  const float cf = __cosf(fr), sf = __sinf(fr);
  const float sv = (i2 + 0.4f * DHh) / (1.4f * DHh);
  const float pw = ((float)n - (float)(Nn / 2)) / 4096.0f;
  const float sc = __powf(sv, pw);

  { // Q: wave w handles head w
    const float* qp = qF + (size_t)bn * (Hh * DHh) + w * DHh + 2 * lane;
    float x0 = qp[0], x1 = qp[1];
    float ss = x0 * x0 + x1 * x1;
#pragma unroll
    for (int m = 16; m >= 1; m >>= 1) ss += __shfl_xor(ss, m);
    float inv = 1.0f / fmaxf(__fsqrt_rn(ss), 1e-12f);
    float g0 = qg[w * DHh + 2 * lane], g1 = qg[w * DHh + 2 * lane + 1];
    x0 = x0 * inv * g0 * 8.0f;   // sqrt(dh) = 8
    x1 = x1 * inv * g1 * 8.0f;
    float r0 = (x0 * cf - x1 * sf) * sc;
    float r1 = (x1 * cf + x0 * sf) * sc;
    __bf16* qo = qh + ((size_t)(b * Hh + w) * Nn + n) * DHh + 2 * lane;
    qo[0] = f2bf(r0); qo[1] = f2bf(r1);
  }
  if (w < KVHh) { // K: waves 0..1 handle kv heads (uniform per wave)
    const float* kp = kvF + (size_t)bn * (KVHh * DHh * 2) + w * DHh + 2 * lane;
    float x0 = kp[0], x1 = kp[1];
    float ss = x0 * x0 + x1 * x1;
#pragma unroll
    for (int m = 16; m >= 1; m >>= 1) ss += __shfl_xor(ss, m);
    float inv = 1.0f / fmaxf(__fsqrt_rn(ss), 1e-12f);
    float g0 = kg[w * DHh + 2 * lane], g1 = kg[w * DHh + 2 * lane + 1];
    x0 = x0 * inv * g0 * 8.0f;
    x1 = x1 * inv * g1 * 8.0f;
    float isc = 1.0f / sc;
    float r0 = (x0 * cf - x1 * sf) * isc;
    float r1 = (x1 * cf + x0 * sf) * isc;
    __bf16* ko = kh + ((size_t)(b * KVHh + w) * Nn + n) * DHh + 2 * lane;
    ko[0] = f2bf(r0); ko[1] = f2bf(r1);
  }
  if (tid < KVHh * DHh) { // V^T: [b, kvh, dh, n]
    int kv = tid / DHh, dh = tid % DHh;
    float v = kvF[(size_t)bn * (KVHh * DHh * 2) + KVHh * DHh + kv * DHh + dh];
    vT[((size_t)(b * KVHh + kv) * DHh + dh) * Nn + n] = f2bf(v);
  }
}

// ---------------------------------------------------------------------------
// Flash attention: grid (Nn/128, Bb*Hh), 256 threads; wave = 16 query rows
// S = (q/sqrt(dh)) @ k^T via WMMA, online softmax, O += P @ V via WMMA
// ---------------------------------------------------------------------------
__global__ __launch_bounds__(256)
void k_flash_attn(const __bf16* __restrict__ qh, const __bf16* __restrict__ kh,
                  const __bf16* __restrict__ vT, __bf16* __restrict__ attnB) {
  __shared__ __align__(16) __bf16 k_lds[64 * 72];
  __shared__ __align__(16) __bf16 v_lds[64 * 72];
  __shared__ __align__(16) __bf16 p_lds[128 * 72];
  const int tid  = threadIdx.x;
  const int lane = tid & 31, w = tid >> 5;
  const int lm = lane & 15, lh = lane >> 4;
  const int bh = blockIdx.y;
  const int b = bh / Hh, h = bh % Hh;
  const int kvh = h % KVHh;                       // GQA tile mapping
  const int qbase = blockIdx.x * 128 + w * 16;

  const __bf16* qptr = qh + (size_t)bh * Nn * DHh;
  const __bf16* kptr = kh + (size_t)(b * KVHh + kvh) * Nn * DHh;
  const __bf16* vptr = vT + (size_t)(b * KVHh + kvh) * DHh * Nn;

  // q fragments resident in registers, pre-scaled by 1/sqrt(dh)
  FragBF qf[2];
#pragma unroll
  for (int s2 = 0; s2 < 2; ++s2) {
    const __bf16* qp = qptr + (size_t)(qbase + lm) * DHh + s2 * 32 + lh * 8;
    qf[s2].q[0] = *(const uint4*)qp;
    qf[s2].q[1] = *(const uint4*)(qp + 16);
#pragma unroll
    for (int e = 0; e < 16; ++e) qf[s2].v[e] = f2bf(bf2f(qf[s2].v[e]) * 0.125f);
  }

  v8f zero;
#pragma unroll
  for (int e = 0; e < 8; ++e) zero[e] = 0.0f;
  v8f O[4];
#pragma unroll
  for (int i = 0; i < 4; ++i) O[i] = zero;
  float m_r[8], l_r[8];
#pragma unroll
  for (int r = 0; r < 8; ++r) { m_r[r] = -3.0e38f; l_r[r] = 0.0f; }

  for (int kb = 0; kb < Nn / 64; ++kb) {
    __syncthreads();
    { // stage 64x64 K tile and 64(dh)x64(keys) V^T tile
      int r = tid >> 2, c = (tid & 3) * 16;
      const __bf16* kg = kptr + (size_t)(kb * 64 + r) * DHh + c;
      *(uint4*)&k_lds[r * 72 + c]     = *(const uint4*)kg;
      *(uint4*)&k_lds[r * 72 + c + 8] = *(const uint4*)(kg + 8);
      const __bf16* vg = vptr + (size_t)r * Nn + kb * 64 + c;
      *(uint4*)&v_lds[r * 72 + c]     = *(const uint4*)vg;
      *(uint4*)&v_lds[r * 72 + c + 8] = *(const uint4*)(vg + 8);
    }
    __syncthreads();

    // S = q @ k^T  (16 q rows x 64 keys per wave)
    v8f st[4];
#pragma unroll
    for (int t = 0; t < 4; ++t) st[t] = zero;
#pragma unroll
    for (int s2 = 0; s2 < 2; ++s2)
#pragma unroll
      for (int t = 0; t < 4; ++t) {
        FragBF fb;
        const __bf16* kp2 = &k_lds[(t * 16 + lm) * 72 + s2 * 32 + lh * 8];
        fb.q[0] = *(const uint4*)kp2;
        fb.q[1] = *(const uint4*)(kp2 + 16);
        st[t] = __builtin_amdgcn_wmma_f32_16x16x32_bf16(
            false, qf[s2].v, false, fb.v, (short)0, st[t], false, false);
      }

    // online softmax (row stats reduced across the 16 lanes holding each row)
    float mb[8];
#pragma unroll
    for (int r = 0; r < 8; ++r)
      mb[r] = fmaxf(fmaxf(st[0][r], st[1][r]), fmaxf(st[2][r], st[3][r]));
#pragma unroll
    for (int r = 0; r < 8; ++r)
#pragma unroll
      for (int msk = 1; msk < 16; msk <<= 1)
        mb[r] = fmaxf(mb[r], __shfl_xor(mb[r], msk));

    float sf[8], rs[8];
#pragma unroll
    for (int r = 0; r < 8; ++r) {
      float mn = fmaxf(m_r[r], mb[r]);
      sf[r] = __expf(m_r[r] - mn);
      m_r[r] = mn;
      rs[r] = 0.0f;
    }
#pragma unroll
    for (int t = 0; t < 4; ++t)
#pragma unroll
      for (int r = 0; r < 8; ++r) {
        float p = __expf(st[t][r] - m_r[r]);
        st[t][r] = p;
        rs[r] += p;
      }
#pragma unroll
    for (int r = 0; r < 8; ++r) {
#pragma unroll
      for (int msk = 1; msk < 16; msk <<= 1) rs[r] += __shfl_xor(rs[r], msk);
      l_r[r] = l_r[r] * sf[r] + rs[r];
    }
#pragma unroll
    for (int dt = 0; dt < 4; ++dt)
#pragma unroll
      for (int r = 0; r < 8; ++r) O[dt][r] *= sf[r];

    // P: C layout -> LDS -> A layout (per-wave private region)
#pragma unroll
    for (int t = 0; t < 4; ++t)
#pragma unroll
      for (int r = 0; r < 8; ++r)
        p_lds[(w * 16 + lh * 8 + r) * 72 + t * 16 + lm] = f2bf(st[t][r]);
    asm volatile("s_wait_dscnt 0" ::: "memory");

    FragBF pf[2];
#pragma unroll
    for (int s2 = 0; s2 < 2; ++s2) {
      const __bf16* pp = &p_lds[(w * 16 + lm) * 72 + s2 * 32 + lh * 8];
      pf[s2].q[0] = *(const uint4*)pp;
      pf[s2].q[1] = *(const uint4*)(pp + 16);
    }
#pragma unroll
    for (int dt = 0; dt < 4; ++dt)
#pragma unroll
      for (int s2 = 0; s2 < 2; ++s2) {
        FragBF fv;
        const __bf16* vp2 = &v_lds[(dt * 16 + lm) * 72 + s2 * 32 + lh * 8];
        fv.q[0] = *(const uint4*)vp2;
        fv.q[1] = *(const uint4*)(vp2 + 16);
        O[dt] = __builtin_amdgcn_wmma_f32_16x16x32_bf16(
            false, pf[s2].v, false, fv.v, (short)0, O[dt], false, false);
      }
  }

  // normalize and store as [b, n, h*dh] bf16 (GEMM-ready A matrix)
#pragma unroll
  for (int dt = 0; dt < 4; ++dt)
#pragma unroll
    for (int r = 0; r < 8; ++r) {
      float val = O[dt][r] / fmaxf(l_r[r], 1e-20f);
      int rown = qbase + lh * 8 + r;
      attnB[((size_t)(b * Nn + rown)) * (Hh * DHh) + h * DHh + dt * 16 + lm] = f2bf(val);
    }
}

// ---------------------------------------------------------------------------
// Host orchestration
// ---------------------------------------------------------------------------
extern "C" void kernel_launch(void* const* d_in, const int* in_sizes, int n_in,
                              void* d_out, int out_size, void* d_ws, size_t ws_size,
                              hipStream_t stream) {
  (void)in_sizes; (void)n_in; (void)out_size; (void)ws_size;
  const float* x   = (const float*)d_in[0];
  const float* Wq  = (const float*)d_in[1];
  const float* Wkv = (const float*)d_in[2];
  const float* qg  = (const float*)d_in[3];
  const float* kg  = (const float*)d_in[4];
  const float* Wo  = (const float*)d_in[5];
  const float* bo  = (const float*)d_in[6];
  const float* W1  = (const float*)d_in[7];
  const float* b1  = (const float*)d_in[8];
  const float* W2  = (const float*)d_in[9];
  const float* b2  = (const float*)d_in[10];
  float* out = (float*)d_out;

  char* ws = (char*)d_ws;
  size_t off = 0;
  auto alloc = [&](size_t bytes) -> void* {
    void* p = ws + off;
    off = (off + bytes + 255) & ~(size_t)255;
    return p;
  };
  float*  xtF   = (float*) alloc((size_t)Mrows * Dd * 4);
  __bf16* xtB   = (__bf16*)alloc((size_t)Mrows * Dd * 2);
  __bf16* WqT   = (__bf16*)alloc((size_t)(Hh * DHh) * Dd * 2);
  __bf16* WkvT  = (__bf16*)alloc((size_t)(KVHh * DHh * 2) * Dd * 2);
  __bf16* WoT   = (__bf16*)alloc((size_t)Dd * (Hh * DHh) * 2);
  __bf16* W1T   = (__bf16*)alloc((size_t)FFf * Dd * 2);
  __bf16* W2T   = (__bf16*)alloc((size_t)Dd * FFf * 2);
  float*  qFb   = (float*) alloc((size_t)Mrows * (Hh * DHh) * 4);
  float*  kvF   = (float*) alloc((size_t)Mrows * (KVHh * DHh * 2) * 4);
  __bf16* qhB   = (__bf16*)alloc((size_t)Bb * Hh * Nn * DHh * 2);
  __bf16* khB   = (__bf16*)alloc((size_t)Bb * KVHh * Nn * DHh * 2);
  __bf16* vTB   = (__bf16*)alloc((size_t)Bb * KVHh * DHh * Nn * 2);
  __bf16* attnB = (__bf16*)alloc((size_t)Mrows * (Hh * DHh) * 2);
  float*  y1F   = (float*) alloc((size_t)Mrows * Dd * 4);
  __bf16* y1B   = (__bf16*)alloc((size_t)Mrows * Dd * 2);
  __bf16* hB    = (__bf16*)alloc((size_t)Mrows * FFf * 2);
  float*  finF  = (float*) alloc((size_t)Mrows * Dd * 4);

  dim3 tb(32, 8);
  // layout prep
  k_x_transpose<<<dim3(Nn / 32, Dd / 32, Bb), tb, 0, stream>>>(x, xtF, xtB);
  k_w_transpose<<<dim3((Hh * DHh) / 32, Dd / 32), tb, 0, stream>>>(Wq, WqT, Dd, Hh * DHh);
  k_w_transpose<<<dim3((KVHh * DHh * 2) / 32, Dd / 32), tb, 0, stream>>>(Wkv, WkvT, Dd, KVHh * DHh * 2);
  k_w_transpose<<<dim3(Dd / 32, (Hh * DHh) / 32), tb, 0, stream>>>(Wo, WoT, Hh * DHh, Dd);
  k_w_transpose<<<dim3(FFf / 32, Dd / 32), tb, 0, stream>>>(W1, W1T, Dd, FFf);
  k_w_transpose<<<dim3(Dd / 32, FFf / 32), tb, 0, stream>>>(W2, W2T, FFf, Dd);

  // Q / KV projections
  k_gemm_bf16<<<dim3((Hh * DHh) / 128, Mrows / 128), 256, 0, stream>>>(
      xtB, WqT, Dd, Hh * DHh, nullptr, 0, nullptr, qFb, nullptr);
  k_gemm_bf16<<<dim3((KVHh * DHh * 2) / 128, Mrows / 128), 256, 0, stream>>>(
      xtB, WkvT, Dd, KVHh * DHh * 2, nullptr, 0, nullptr, kvF, nullptr);

  // RMSNorm + rotary + V^T
  k_qkv_post<<<Bb * Nn, 256, 0, stream>>>(qFb, kvF, qg, kg, qhB, khB, vTB);

  // flash attention
  k_flash_attn<<<dim3(Nn / 128, Bb * Hh), 256, 0, stream>>>(qhB, khB, vTB, attnB);

  // out projection + residual (f32 + bf16 outputs)
  k_gemm_bf16<<<dim3(Dd / 128, Mrows / 128), 256, 0, stream>>>(
      attnB, WoT, Hh * DHh, Dd, bo, 0, xtF, y1F, y1B);

  // FFN
  k_gemm_bf16<<<dim3(FFf / 128, Mrows / 128), 256, 0, stream>>>(
      y1B, W1T, Dd, FFf, b1, 1, nullptr, nullptr, hB);
  k_gemm_bf16<<<dim3(Dd / 128, Mrows / 128), 256, 0, stream>>>(
      hB, W2T, FFf, Dd, b2, 0, y1F, finF, nullptr);

  // back to [B, D, N]
  k_out_transpose<<<dim3(Dd / 32, Nn / 32, Bb), tb, 0, stream>>>(finF, out);
}
